// reconstruct_Decoder_25786983645635
// MI455X (gfx1250) — matrix operand order repaired and checked by reading, not compiled, not run on hardware
//
#include <hip/hip_runtime.h>
#include <hip/hip_bf16.h>

typedef __attribute__((ext_vector_type(16))) __bf16 v16bf;
typedef __attribute__((ext_vector_type(8)))  float  v8f;

#define NB      64      // batch
#define SEQ     48
#define TSTEPS  47
#define EH      128     // ENC_HID
#define DH      256     // DEC_HID
#define LL      144     // SEQ*3 keys

// packed-weight tile table (per attribute), tiles of 512 bf16 (16x16x32 B-frag order)
#define TILE_WS     0      // attn Ws   [256x256]  16*8
#define TILE_WENC   128    // attn Wenc [256x256]  16*8
#define TILE_WIH0   256    // W_ih0 [768x384] 48*12
#define TILE_WHH0   832    // W_hh0 [768x256] 48*8
#define TILE_WIH1   1216   // W_ih1 [768x256] 48*8
#define TILE_WHH1   1600   // W_hh1 [768x256] 48*8
#define TILE_FC     1984   // fcW [pad64 x 640] 4*20
#define TILES_PER_ATTR 2064
#define PACK_PER_ATTR  ((size_t)TILES_PER_ATTR * 512)

struct PtrTab { const float* p[3][13]; };

static __device__ __forceinline__ v8f wmma_bf16(v16bf a, v16bf b, v8f c) {
  return __builtin_amdgcn_wmma_f32_16x16x32_bf16(false, a, false, b, (short)0, c, false, false);
}

// B fragment: one contiguous 32B blob per lane (pre-packed layout)
static __device__ __forceinline__ v16bf load_b_frag(const __bf16* pk, int tile, int lane) {
  return *(const v16bf*)(pk + (size_t)tile * 512 + lane * 16);
}

// A fragment from LDS f32, row-major. rowbase = &A[row_of_this_lane][0].
// 16-bit A 16x32 layout: lanes 0-15 K in [kb, kb+8)U[16+kb,16+kb+8) with kb=0; lanes 16-31 kb=8.
static __device__ __forceinline__ v16bf a_frag_lds(const float* rowbase, int k0, int lane) {
  int hl = lane >> 4;
  int kb = k0 + hl * 8;
  v16bf a;
#pragma unroll
  for (int g = 0; g < 8; ++g) {
    int kk = (g < 4) ? (kb + 2 * g) : (16 + kb + 2 * (g - 4));
    float2 f = *(const float2*)(rowbase + kk);
    a[2 * g]     = (__bf16)f.x;
    a[2 * g + 1] = (__bf16)f.y;
  }
  return a;
}

// A fragment from global bf16 row-major
static __device__ __forceinline__ v16bf a_frag_g(const __bf16* rowbase, int k0, int lane) {
  int hl = lane >> 4;
  int kb = k0 + hl * 8;
  v16bf a;
#pragma unroll
  for (int g = 0; g < 8; ++g) {
    int kk = (g < 4) ? (kb + 2 * g) : (16 + kb + 2 * (g - 4));
    a[2 * g]     = rowbase[kk];
    a[2 * g + 1] = rowbase[kk + 1];
  }
  return a;
}

// ---------------- weight repack: f32 -> bf16 fragment-order tiles ----------------
__global__ void k_repack(PtrTab pt, __bf16* packed) {
  const int matN[7]      = {256, 256, 768, 768, 768, 768, 64};
  const int matKT[7]     = {8, 8, 12, 8, 8, 8, 20};
  const int matStride[7] = {512, 512, 384, 256, 256, 256, 640};
  const int matColOff[7] = {0, 256, 0, 0, 0, 0, 0};
  const int matSrc[7]    = {0, 0, 4, 5, 8, 9, 2};
  const int cum[8]       = {0, 128, 256, 832, 1216, 1600, 1984, 2064};
  const int OUTD[3]      = {51, 31, 21};

  int wid  = blockIdx.x * 8 + (threadIdx.x >> 5);
  int lane = threadIdx.x & 31;
  if (wid >= 3 * TILES_PER_ATTR) return;
  int attr = wid / TILES_PER_ATTR, tl = wid % TILES_PER_ATTR;
  int m = 0;
  while (tl >= cum[m + 1]) ++m;
  int lt = tl - cum[m];
  int KT = matKT[m];
  int nbT = lt / KT, kbT = lt % KT;
  int lrow = lane & 15, hl = lane >> 4;
  int n = nbT * 16 + lrow;
  int realN = (m == 6) ? OUTD[attr] : matN[m];
  const float* src = pt.p[attr][matSrc[m]];
  v16bf vv;
#pragma unroll
  for (int j = 0; j < 16; ++j) {
    int k = kbT * 32 + ((j < 8) ? (hl * 8 + j) : (16 + hl * 8 + (j - 8)));
    float val = (n < realN) ? src[n * matStride[m] + matColOff[m] + k] : 0.0f;
    vv[j] = (__bf16)val;
  }
  *(v16bf*)(packed + (size_t)attr * PACK_PER_ATTR + (size_t)tl * 512 + lane * 16) = vv;
}

// ---------------- enc_output f32 -> bf16 ----------------
__global__ void k_encbf(const float* enc, __bf16* encbf) {
  int idx = blockIdx.x * 256 + threadIdx.x;
  if (idx < LL * NB * 256) encbf[idx] = (__bf16)enc[idx];
}

// ---------------- enc_proj[a][l][d][b] = (enc[l] @ Wenc.T)^T, bf16, WMMA ----------------
__global__ __launch_bounds__(256) void k_encproj(const __bf16* encbf, const __bf16* packed,
                                                 __bf16* encproj) {
  int wid  = blockIdx.x * 8 + (threadIdx.x >> 5);
  int lane = threadIdx.x & 31;
  int lrow = lane & 15, hl = lane >> 4;
  int nb = wid & 15, mb = (wid >> 4) & 3;
  int l = (wid >> 6) % LL, attr = wid / (64 * LL);
  if (attr >= 3) return;
  const __bf16* pk = packed + (size_t)attr * PACK_PER_ATTR;
  const __bf16* arow = encbf + (size_t)(l * NB + mb * 16 + lrow) * 256;
  v8f acc = {};
#pragma unroll
  for (int kb = 0; kb < 8; ++kb) {
    acc = wmma_bf16(a_frag_g(arow, kb * 32, lane),
                    load_b_frag(pk, TILE_WENC + nb * 8 + kb, lane), acc);
  }
  int n = nb * 16 + lrow;
  __bf16* op = encproj + ((size_t)(attr * LL + l) * 256 + n) * NB;
#pragma unroll
  for (int g = 0; g < 8; ++g) op[mb * 16 + g + hl * 8] = (__bf16)acc[g];
}

// ---------------- step-0 one-hot quirk ----------------
__global__ void k_inithot(const int* Xs0, const int* Xs1, const int* Xs2, float* out) {
  int idx = blockIdx.x * 256 + threadIdx.x;
  if (idx >= NB * 103) return;
  int b = idx / 103, c = idx % 103;
  int o, off, outd, X;
  if (c < 51)      { o = c;      off = 0;                       outd = 51; X = Xs0[0]; }
  else if (c < 82) { o = c - 51; off = 3072 * 51;               outd = 31; X = Xs1[0]; }
  else             { o = c - 82; off = 3072 * 51 + 3072 * 31;   outd = 21; X = Xs2[0]; }
  out[off + (b * SEQ) * outd + o] = (o == X) ? 1.0f : 0.0f;
}

// ---------------- one GRU layer, WMMA, results staged in registers across barrier ----------------
static __device__ __forceinline__ void gru_layer(const float* xbase, int xstride, int KTx,
                                                 int tile_ih, const float* sh_s, int tile_hh,
                                                 const __bf16* pk, const float* bih,
                                                 const float* bhh, float* dest, int w, int lane) {
  int lrow = lane & 15, hl = lane >> 4;
  int mb = w >> 3;
  int nbb = (w & 7) * 2;
  float hreg[2][8];
#pragma unroll
  for (int j = 0; j < 2; ++j) {
    int nb = nbb + j;
    v8f aR = {}, aZ = {}, aN = {}, aH = {};
    for (int kb = 0; kb < KTx; ++kb) {                 // x side: r,z,i_n gates
      v16bf af = a_frag_lds(xbase + (mb * 16 + lrow) * xstride, kb * 32, lane);
      aR = wmma_bf16(af, load_b_frag(pk, tile_ih + nb * KTx + kb, lane), aR);
      aZ = wmma_bf16(af, load_b_frag(pk, tile_ih + (16 + nb) * KTx + kb, lane), aZ);
      aN = wmma_bf16(af, load_b_frag(pk, tile_ih + (32 + nb) * KTx + kb, lane), aN);
    }
    for (int kb = 0; kb < 8; ++kb) {                   // hidden side: r,z fused, h_n separate
      v16bf af = a_frag_lds(sh_s + (mb * 16 + lrow) * 256, kb * 32, lane);
      aR = wmma_bf16(af, load_b_frag(pk, tile_hh + nb * 8 + kb, lane), aR);
      aZ = wmma_bf16(af, load_b_frag(pk, tile_hh + (16 + nb) * 8 + kb, lane), aZ);
      aH = wmma_bf16(af, load_b_frag(pk, tile_hh + (32 + nb) * 8 + kb, lane), aH);
    }
    int n = nb * 16 + lrow;
    float bR = bih[n] + bhh[n];
    float bZ = bih[256 + n] + bhh[256 + n];
    float bIN = bih[512 + n], bHN = bhh[512 + n];
#pragma unroll
    for (int g = 0; g < 8; ++g) {
      int m = mb * 16 + g + hl * 8;
      float r  = 1.0f / (1.0f + __expf(-(aR[g] + bR)));
      float z  = 1.0f / (1.0f + __expf(-(aZ[g] + bZ)));
      float nn = tanhf(aN[g] + bIN + r * (aH[g] + bHN));
      hreg[j][g] = (1.0f - z) * nn + z * sh_s[m * 256 + n];
    }
  }
  __syncthreads();   // all reads of sh_s / xbase done before dest (may alias sh_s) is written
#pragma unroll
  for (int j = 0; j < 2; ++j) {
    int n = (nbb + j) * 16 + lrow;
#pragma unroll
    for (int g = 0; g < 8; ++g) dest[(mb * 16 + g + hl * 8) * 256 + n] = hreg[j][g];
  }
  __syncthreads();
}

static __device__ __forceinline__ float fc_in(const float* sh_s, const float* sh_x, int m, int k) {
  if (k < 256) return sh_s[m * 256 + k];                 // dec_output (new s)
  if (k < 512) return sh_x[m * 384 + 128 + (k - 256)];   // c
  return sh_x[m * 384 + (k - 512)];                      // emb
}

// ---------------- persistent decoder: 1 workgroup (1024 thr) per attribute ----------------
__global__ __launch_bounds__(1024, 1)
void k_decode(PtrTab pt, const int* Xs0, const unsigned char* mask, const float* s_in,
              const __bf16* encbf, const __bf16* encproj, const __bf16* packed, float* out) {
  extern __shared__ float sm[];
  float* sh_s  = sm;                  // [64][256] current hidden s (f32)
  float* sh_h  = sh_s + NB * 256;     // [d][b] s_proj^T, then [64][256] h1
  float* sh_x  = sh_h + NB * 256;     // [64][384] = [emb(128) | c(256)]
  float* sh_sc = sh_x + NB * 384;     // [64][144] scores -> probs
  float* sh_v  = sh_sc + NB * LL;     // [256]

  const int a = blockIdx.x;
  const int tid = threadIdx.x, lane = tid & 31, w = tid >> 5;
  const int lrow = lane & 15, hl = lane >> 4;

  const int   OUTD[3] = {51, 31, 21};
  const int   OOFF[3] = {0, 3072 * 51, 3072 * 51 + 3072 * 31};
  const int   outd = OUTD[a];
  float*      outp = out + OOFF[a];
  const float* embT = pt.p[a][1];
  const float* fcb  = pt.p[a][3];
  const float* bih0 = pt.p[a][6],  *bhh0 = pt.p[a][7];
  const float* bih1 = pt.p[a][10], *bhh1 = pt.p[a][11];
  const float* vvec = pt.p[a][12];
  const __bf16* pk = packed + (size_t)a * PACK_PER_ATTR;

  for (int i = tid; i < NB * 256; i += 1024) sh_s[i] = s_in[a * NB * 256 + i];
  for (int i = tid; i < 256; i += 1024) sh_v[i] = vvec[i];
  __syncthreads();

  for (int t = 0; t < TSTEPS; ++t) {
    // ---- P0: embedding lookup into sh_x[:, 0:128]
    int tcol = (a == 0) ? t : (t + 1);
#pragma unroll
    for (int i = 0; i < 8; ++i) {
      int idx = i * 1024 + tid, b = idx >> 7, e = idx & 127;
      int tok = Xs0[b * SEQ + tcol];
      tok = (tok < 0) ? 0 : ((tok > 50) ? 50 : tok);
      sh_x[b * 384 + e] = embT[tok * 128 + e];
    }
    // ---- P1: s_proj^T = (s @ Ws.T)^T into sh_h[d][b]   (WMMA)
    {
      int mb = w >> 3, nbb = (w & 7) * 2;
#pragma unroll
      for (int j = 0; j < 2; ++j) {
        int nb = nbb + j;
        v8f acc = {};
#pragma unroll
        for (int kb = 0; kb < 8; ++kb)
          acc = wmma_bf16(a_frag_lds(sh_s + (mb * 16 + lrow) * 256, kb * 32, lane),
                          load_b_frag(pk, TILE_WS + nb * 8 + kb, lane), acc);
        int n = nb * 16 + lrow;
#pragma unroll
        for (int g = 0; g < 8; ++g) sh_h[n * NB + mb * 16 + g + hl * 8] = acc[g];
      }
    }
    __syncthreads();
    // ---- P2: scores[b][l] = v . tanh(enc_proj + s_proj), masked
#pragma unroll 1
    for (int i = 0; i < 9; ++i) {
      int idx = i * 1024 + tid, l = idx >> 6, b = idx & 63;
      const __bf16* ep = encproj + (size_t)(a * LL + l) * 256 * NB + b;
      float acc = 0.0f;
      for (int d = 0; d < 256; ++d)
        acc += sh_v[d] * tanhf((float)ep[(size_t)d * NB] + sh_h[d * NB + b]);
      bool mk = mask[b * SEQ + (l % SEQ)] != 0;
      sh_sc[b * LL + l] = mk ? -1e9f : acc;
    }
    __syncthreads();
    // ---- P3: softmax over L per batch row (wave handles 2 rows, shuffle reductions)
#pragma unroll
    for (int j = 0; j < 2; ++j) {
      int b = w * 2 + j;
      float mx = -3.4e38f;
      for (int l = lane; l < LL; l += 32) mx = fmaxf(mx, sh_sc[b * LL + l]);
      for (int o = 16; o; o >>= 1) mx = fmaxf(mx, __shfl_xor(mx, o, 32));
      float sum = 0.0f;
      for (int l = lane; l < LL; l += 32) sum += __expf(sh_sc[b * LL + l] - mx);
      for (int o = 16; o; o >>= 1) sum += __shfl_xor(sum, o, 32);
      float inv = 1.0f / sum;
      for (int l = lane; l < LL; l += 32)
        sh_sc[b * LL + l] = __expf(sh_sc[b * LL + l] - mx) * inv;
    }
    __syncthreads();
    // ---- P4: context c[b][e] = sum_l probs * enc (coalesced bf16, prefetch ahead)
#pragma unroll 1
    for (int i = 0; i < 16; ++i) {
      int tt = i * 32 + w, b = tt >> 3, eb = tt & 7, e = eb * 32 + lane;
      const __bf16* ep = encbf + (size_t)b * 256 + e;
      float acc = 0.0f;
      for (int l = 0; l < LL; ++l) {
        if ((l & 15) == 0 && l + 16 < LL)
          __builtin_prefetch(ep + (size_t)(l + 16) * NB * 256, 0, 1);
        acc += sh_sc[b * LL + l] * (float)ep[(size_t)l * NB * 256];
      }
      sh_x[b * 384 + 128 + e] = acc;
    }
    __syncthreads();
    // ---- P5/P6: two GRU layers (both use s as hidden; h carried x->h1->h2)
    gru_layer(sh_x, 384, 12, TILE_WIH0, sh_s, TILE_WHH0, pk, bih0, bhh0, sh_h, w, lane);
    gru_layer(sh_h, 256, 8,  TILE_WIH1, sh_s, TILE_WHH1, pk, bih1, bhh1, sh_s, w, lane);
    // ---- P7: fc head, pred = [h2|c|emb] @ fcW.T + fcb  (WMMA, 16 tiles on 16 waves)
    if (w < 16) {
      int mb = w >> 2, nb = w & 3;
      v8f acc = {};
#pragma unroll 1
      for (int kb = 0; kb < 20; ++kb) {
        v16bf af;
        int kbb = kb * 32 + hl * 8;
        int m = mb * 16 + lrow;
#pragma unroll
        for (int g = 0; g < 8; ++g) {
          int kk = (g < 4) ? (kbb + 2 * g) : (16 + kbb + 2 * (g - 4));
          af[2 * g]     = (__bf16)fc_in(sh_s, sh_x, m, kk);
          af[2 * g + 1] = (__bf16)fc_in(sh_s, sh_x, m, kk + 1);
        }
        acc = wmma_bf16(af, load_b_frag(pk, TILE_FC + nb * 20 + kb, lane), acc);
      }
      int n = nb * 16 + lrow;
      if (n < outd) {
        float bb = fcb[n];
#pragma unroll
        for (int g = 0; g < 8; ++g) {
          int m = mb * 16 + g + hl * 8;
          outp[(m * SEQ + (t + 1)) * outd + n] = acc[g] + bb;
        }
      }
    }
    __syncthreads();   // sh_x / sh_s stable before next step overwrites
  }
}

extern "C" void kernel_launch(void* const* d_in, const int* in_sizes, int n_in,
                              void* d_out, int out_size, void* d_ws, size_t ws_size,
                              hipStream_t stream) {
  (void)in_sizes; (void)n_in; (void)out_size; (void)ws_size;
  const int* Xs0 = (const int*)d_in[0];
  const int* Xs1 = (const int*)d_in[1];
  const int* Xs2 = (const int*)d_in[2];
  const float* s_in = (const float*)d_in[3];
  const float* enc  = (const float*)d_in[4];
  const unsigned char* mask = (const unsigned char*)d_in[5];
  PtrTab pt;
  for (int a = 0; a < 3; ++a)
    for (int j = 0; j < 13; ++j)
      pt.p[a][j] = (const float*)d_in[6 + a * 13 + j];

  char* ws = (char*)d_ws;
  __bf16* packed = (__bf16*)ws;                                    // 3*2064*512*2 B
  size_t off = (3 * PACK_PER_ATTR * 2 + 255) & ~(size_t)255;
  __bf16* encbf = (__bf16*)(ws + off);                             // 144*64*256*2 B
  off = (off + (size_t)LL * NB * 256 * 2 + 255) & ~(size_t)255;
  __bf16* encproj = (__bf16*)(ws + off);                           // 3*144*256*64*2 B

  k_repack<<<774, 256, 0, stream>>>(pt, packed);
  k_encbf<<<(LL * NB * 256 + 255) / 256, 256, 0, stream>>>(enc, encbf);
  k_encproj<<<3 * LL * 64 / 8, 256, 0, stream>>>(encbf, packed, encproj);
  k_inithot<<<(NB * 103 + 255) / 256, 256, 0, stream>>>(Xs0, Xs1, Xs2, (float*)d_out);

  size_t shmem = (size_t)(NB * 256 + NB * 256 + NB * 384 + NB * LL + 256) * sizeof(float);
  (void)hipFuncSetAttribute((const void*)k_decode,
                            hipFuncAttributeMaxDynamicSharedMemorySize, (int)shmem);
  k_decode<<<3, 1024, shmem, stream>>>(pt, Xs0, mask, s_in, encbf, encproj, packed,
                                       (float*)d_out);
}
